// Brain_69939247448536
// MI455X (gfx1250) — matrix operand order repaired and checked by reading, not compiled, not run on hardware
//
#include <hip/hip_runtime.h>
#include <hip/hip_bf16.h>
#include <stdint.h>

// ---------------------------------------------------------------------------
// CDNA5 (gfx1250) wave32 WMMA helpers
// ---------------------------------------------------------------------------
typedef __attribute__((ext_vector_type(16))) __bf16 v16bf;
typedef __attribute__((ext_vector_type(8)))  float  v8f;

__device__ __forceinline__ unsigned short bf_bits(float f) {
    return __builtin_bit_cast(unsigned short, (__bf16)f);
}
__device__ __forceinline__ __bf16 bits_to_bf16(unsigned short b) {
    return __builtin_bit_cast(__bf16, b);
}
__device__ __forceinline__ float bf2f(unsigned short b) {
    unsigned int u = ((unsigned int)b) << 16;
    return __builtin_bit_cast(float, u);
}
__device__ __forceinline__ v8f wmma_bf16(v16bf a, v16bf b, v8f c) {
    // D(f32 16x16) = A(bf16 16x32) * B(bf16 32x16) + C
    return __builtin_amdgcn_wmma_f32_16x16x32_bf16(
        false, a, false, b, (short)0, c, false, false);
}
// Contiguous-k element index inside a 16x32 A tile (this code's order):
// lane = m + 16*(k>>4), elem = k&15
__device__ __forceinline__ int a_swz(int m, int k) {
    return (m + 16 * (k >> 4)) * 16 + (k & 15);
}
// Contiguous-k element index inside a 32x16 B tile (same order as A):
// lane = n + 16*(k>>4), elem = k&15
__device__ __forceinline__ int b_swz(int k, int n) {
    return (n + 16 * (k >> 4)) * 16 + (k & 15);
}
// Global fragment-swizzled layouts, tile-major:
//   A matrix [M,W]:  tile (mt,kt) at ((mt*(W/32)+kt)*512)
//   B matrix [K,N]:  tile (nt,kt) at ((nt*(K/32)+kt)*512)
__device__ __forceinline__ size_t aswz_g(int m, int k, int W) {
    return ((size_t)((m >> 4) * (W >> 5) + (k >> 5))) * 512 + a_swz(m & 15, k & 31);
}
__device__ __forceinline__ size_t bswz_g(int k, int n, int K) {
    return ((size_t)((n >> 4) * (K >> 5) + (k >> 5))) * 512 + b_swz(k & 31, n & 15);
}

// ---------------------------------------------------------------------------
// Conversion kernels (write fragment-swizzled global layouts)
// ---------------------------------------------------------------------------
// B = src^T (src is [N rows, K cols] fp32) -> B-swizzled [K,N]
__global__ void k_swizzleBT_f32_bf16(const float* __restrict__ src,
                                     unsigned short* __restrict__ dst,
                                     int K, int N) {
    int i = blockIdx.x * blockDim.x + threadIdx.x;
    if (i >= K * N) return;
    int k = i / N, n = i - k * N;
    dst[bswz_g(k, n, K)] = bf_bits(src[(size_t)n * K + k]);
}
// B = src (src is [K rows, N cols] fp32) -> B-swizzled [K,N]
__global__ void k_swizzleB_f32_bf16(const float* __restrict__ src,
                                    unsigned short* __restrict__ dst,
                                    int K, int N) {
    int i = blockIdx.x * blockDim.x + threadIdx.x;
    if (i >= K * N) return;
    int k = i / N, n = i - k * N;
    dst[bswz_g(k, n, K)] = bf_bits(src[i]);
}
// fp32 [rows, sld] -> A-swizzled bf16 submatrix of width obW at column obCol
__global__ void k_conv_aswz(const float* __restrict__ src, int sld,
                            unsigned short* __restrict__ dst,
                            int obCol, int obW, int rows, int cols) {
    int i = blockIdx.x * blockDim.x + threadIdx.x;
    if (i >= rows * cols) return;
    int r = i / cols, c = i - r * cols;
    dst[aswz_g(r, obCol + c, obW)] = bf_bits(src[(size_t)r * sld + c]);
}
// embedding gather -> A-swizzled bf16 [rows, D]
__global__ void k_embed_bf16(const int* __restrict__ x,
                             const float* __restrict__ soma,
                             unsigned short* __restrict__ dst,
                             int rows, int D) {
    int i = blockIdx.x * blockDim.x + threadIdx.x;
    if (i >= rows * D) return;
    int r = i / D, d = i - r * D;
    dst[aswz_g(r, d, D)] = bf_bits(soma[(size_t)x[r] * D + d]);
}
// row-normalize soma_w, write B-swizzled sw^T [D,V]
__global__ __launch_bounds__(256) void k_norm_soma(const float* __restrict__ soma,
                                                   unsigned short* __restrict__ swT,
                                                   int V, int D) {
    __shared__ float red[256];
    int v = blockIdx.x, tid = threadIdx.x;
    float s = 0.f;
    for (int d = tid; d < D; d += 256) { float w = soma[(size_t)v * D + d]; s += w * w; }
    red[tid] = s; __syncthreads();
    for (int o = 128; o > 0; o >>= 1) { if (tid < o) red[tid] += red[tid + o]; __syncthreads(); }
    float sc = 1.f / fmaxf(sqrtf(red[0]), 1e-12f);
    for (int d = tid; d < D; d += 256)
        swT[bswz_g(d, v, D)] = bf_bits(soma[(size_t)v * D + d] * sc);
}

// ---------------------------------------------------------------------------
// bf16 WMMA GEMM: C[M,N] = act(scale * A@B + bias).  Requires M%64==0,
// N%64==0, K%32==0.  A/B pre-swizzled tile-major in global; staging is one
// GLOBAL_LOAD_ASYNC_TO_LDS_B128 per thread per operand per k-step, double
// buffered so the async DMA of block i+1 overlaps the WMMAs of block i.
// actCode: 0 none, 1 elu(x)+1 on cols < actLim, 2 tanh
// ---------------------------------------------------------------------------
__global__ __launch_bounds__(256) void k_gemm_bf16(
    const unsigned short* __restrict__ A, const unsigned short* __restrict__ B,
    const float* __restrict__ bias,
    float* outF, int ldF, unsigned short* outB, int obCol, int obW,
    int M, int N, int K, float scale, int actCode, int actLim) {
    __shared__ __align__(32) unsigned short As[2][4 * 512];
    __shared__ __align__(32) unsigned short Bs[2][4 * 512];
    int tid = threadIdx.x, lane = tid & 31, wid = tid >> 5, half = lane >> 4;
    int mi = wid & 3, nj = wid >> 2;                 // 4x2 wave grid, 2 n-tiles/wave
    int m0 = blockIdx.y * 64, n0 = blockIdx.x * 64;
    int kt_per = K >> 5;
    int ti = tid >> 6;                               // tile 0..3 staged by this thread
    int off = (tid & 63) << 3;                       // 8 u16 = 16B per thread

    auto stage = [&](int buf, int k0) {
        int kt0 = k0 >> 5;
        const unsigned short* ga = A + ((size_t)((m0 >> 4) + ti) * kt_per + kt0) * 512 + off;
        const unsigned short* gb = B + ((size_t)((n0 >> 4) + ti) * kt_per + kt0) * 512 + off;
        unsigned la = (unsigned)(uintptr_t)(&As[buf][ti * 512 + off]);
        unsigned lb = (unsigned)(uintptr_t)(&Bs[buf][ti * 512 + off]);
        asm volatile("global_load_async_to_lds_b128 %0, %1, off"
                     :: "v"(la), "v"(ga) : "memory");
        asm volatile("global_load_async_to_lds_b128 %0, %1, off"
                     :: "v"(lb), "v"(gb) : "memory");
    };

    v8f c0 = {}; v8f c1 = {};
    stage(0, 0);
    asm volatile("s_wait_asynccnt 0" ::: "memory");
    __syncthreads();
    int buf = 0;
    for (int k0 = 0; k0 < K; k0 += 32) {
        if (k0 + 32 < K) stage(buf ^ 1, k0 + 32);    // async prefetch next block
        v16bf af = *(const v16bf*)(&As[buf][mi * 512 + lane * 16]);
        v16bf b0 = *(const v16bf*)(&Bs[buf][(nj * 2 + 0) * 512 + lane * 16]);
        v16bf b1 = *(const v16bf*)(&Bs[buf][(nj * 2 + 1) * 512 + lane * 16]);
        c0 = wmma_bf16(af, b0, c0);
        c1 = wmma_bf16(af, b1, c1);
        asm volatile("s_wait_asynccnt 0" ::: "memory");
        __syncthreads();
        buf ^= 1;
    }
    #pragma unroll
    for (int u = 0; u < 2; ++u) {
        v8f c = u ? c1 : c0;
        int nt = nj * 2 + u;
        int gn = n0 + nt * 16 + (lane & 15);
        #pragma unroll
        for (int j = 0; j < 8; ++j) {
            int gm = m0 + mi * 16 + 8 * half + j;
            float v = c[j] * scale;
            if (bias) v += bias[gn];
            if (actCode == 1 && gn < actLim) v = (v > 0.f) ? (v + 1.f) : __expf(v);
            else if (actCode == 2) v = tanhf(v);
            if (outF) outF[(size_t)gm * ldF + gn] = v;
            if (outB) outB[aswz_g(gm, obCol + gn, obW)] = bf_bits(v);
        }
    }
}

// ---------------------------------------------------------------------------
// Sequential GRU: one persistent workgroup; h in LDS (fp32 master + bf16
// A-fragment-swizzled tile).  Whh^T pre-swizzled tile-major in global ->
// per-step B fragments are coalesced 32B vector loads.
// ---------------------------------------------------------------------------
__global__ __launch_bounds__(256) void k_gru_seq(
    const float* __restrict__ GI,             // [B*T, 1536] = x@Wih^T + bih
    const unsigned short* __restrict__ Wsw,   // bf16 swizzled [(nt*16+kt)*512]
    const float* __restrict__ bhh,            // [1536]
    const float* __restrict__ h0,             // [B, 512]
    float* __restrict__ P,                    // [B*T, 512] fp32
    unsigned short* __restrict__ Pb,          // bf16 A-swizzled [B*T, 512]
    float* __restrict__ hT) {                 // [B, 512]
    const int G = 1536;
    __shared__ float hF[2][512];
    __shared__ __align__(32) unsigned short hB[16 * 512];   // 16 kt-tiles, swizzled
    __shared__ float gh[2][1536];
    int tid = threadIdx.x, lane = tid & 31, wid = tid >> 5, half = lane >> 4;
    for (int i = tid; i < 16 * 512; i += 256) {
        int r = i >> 9, d = i & 511;
        float v = (r < 2) ? h0[r * 512 + d] : 0.f;
        hB[(d >> 5) * 512 + a_swz(r, d & 31)] = bf_bits(v);
        if (r < 2) hF[r][d] = v;
    }
    __syncthreads();
    for (int t = 0; t < 512; ++t) {
        // gh[0:2,1536] = h @ Whh^T   (96 N-tiles, 12 per wave)
        for (int i = 0; i < 12; ++i) {
            int nt = wid * 12 + i;
            v8f acc = {};
            for (int kt = 0; kt < 16; ++kt) {
                v16bf af = *(const v16bf*)(hB + kt * 512 + lane * 16);
                v16bf bfr = *(const v16bf*)(Wsw + ((size_t)(nt * 16 + kt)) * 512 + lane * 16);
                acc = wmma_bf16(af, bfr, acc);
            }
            int bn = nt * 16 + (lane & 15);
            #pragma unroll
            for (int j = 0; j < 8; ++j) {
                int m = 8 * half + j;
                if (m < 2) gh[m][bn] = acc[j];
            }
        }
        __syncthreads();
        // gate math (torch GRU order r,z,n)
        for (int u = tid; u < 1024; u += 256) {
            int b = u >> 9, d = u & 511;
            int row = (b << 9) | t;
            const float* gi = GI + (size_t)row * G;
            float hr = gh[b][d] + bhh[d];
            float hz = gh[b][512 + d] + bhh[512 + d];
            float hn = gh[b][1024 + d] + bhh[1024 + d];
            float r = 1.f / (1.f + __expf(-(gi[d] + hr)));
            float z = 1.f / (1.f + __expf(-(gi[512 + d] + hz)));
            float n = tanhf(gi[1024 + d] + r * hn);
            float hv = (1.f - z) * n + z * hF[b][d];
            hF[b][d] = hv;
            hB[(d >> 5) * 512 + a_swz(b, d & 31)] = bf_bits(hv);
            P[(size_t)row * 512 + d] = hv;
            Pb[aswz_g(row, d, 512)] = bf_bits(hv);
        }
        __syncthreads();
    }
    for (int u = tid; u < 1024; u += 256) hT[u] = hF[u >> 9][u & 511];
}

// ---------------------------------------------------------------------------
// Chunked causal linear attention (chunk = 16 = WMMA M).
// q staged A-swizzled; v staged compact B-swizzled (k<16 half only);
// state S kept in C-fragment-native 16x16 tile order in workspace so the
// rank-16 update C load/store and the q@S B-fragments are coalesced b128s.
// ---------------------------------------------------------------------------
__global__ __launch_bounds__(256) void k_linattn(
    const float* __restrict__ QKV,            // [B*T, 1536]: q',k' (elu+1), v
    float* __restrict__ S,                    // [B] x 32x32 tiles of 16x16 f32
    unsigned short* __restrict__ outB, int obCol, int obW) {
    const int D = 512, G = 1536;
    __shared__ __align__(32) unsigned short qs[16 * 512];   // A-swizzled kt-tiles
    __shared__ unsigned short ks[16][512];                  // plain [t][d]
    __shared__ __align__(32) unsigned short vsw[32 * 256];  // compact B-swizzled
    __shared__ float att[16][32];                           // cols 16..31 = 0
    __shared__ float ksum[512];
    __shared__ float den[16];
    int tid = threadIdx.x, lane = tid & 31, wid = tid >> 5, half = lane >> 4;
    int b = blockIdx.x;
    float* Sb = S + (size_t)b * D * D;
    v16bf vzero;
    #pragma unroll
    for (int e = 0; e < 16; ++e) vzero[e] = (__bf16)0.f;
    for (int i = tid; i < D * D; i += 256) Sb[i] = 0.f;
    for (int i = tid; i < 512; i += 256) ksum[i] = 0.f;
    __syncthreads();
    for (int c = 0; c < 32; ++c) {
        int rowBase = b * 512 + c * 16;
        for (int i = tid; i < 16 * 512; i += 256) {
            int r = i >> 9, d = i & 511;
            const float* src = QKV + (size_t)(rowBase + r) * G;
            qs[(d >> 5) * 512 + a_swz(r, d & 31)] = bf_bits(src[d]);
            ks[r][d] = bf_bits(src[512 + d]);
            vsw[(d >> 4) * 256 + (d & 15) * 16 + r] = bf_bits(src[1024 + d]);
        }
        for (int i = tid; i < 16 * 32; i += 256) att[i >> 5][i & 31] = 0.f;
        __syncthreads();
        // att[16,16] = q_chunk @ k_chunk^T   (K=512 split over 8 waves)
        for (int kk2 = 0; kk2 < 2; ++kk2) {
            int kt = wid * 2 + kk2;
            v16bf af = *(const v16bf*)(qs + kt * 512 + lane * 16);
            v16bf bfr;
            int bn = lane & 15;
            #pragma unroll
            for (int e = 0; e < 16; ++e)
                bfr[e] = bits_to_bf16(ks[bn][kt * 32 + 16 * half + e]);
            v8f pc = {};
            pc = wmma_bf16(af, bfr, pc);
            #pragma unroll
            for (int j = 0; j < 8; ++j) atomicAdd(&att[8 * half + j][bn], pc[j]);
        }
        __syncthreads();
        // denominator from att row sums + q . ksum_prev, then causal mask
        if (tid < 16) {
            int t = tid;
            float dq = 0.f;
            for (int d = 0; d < 512; ++d)
                dq += bf2f(qs[(d >> 5) * 512 + a_swz(t, d & 31)]) * ksum[d];
            float rs = 0.f;
            for (int s = 0; s <= t; ++s) rs += att[t][s];
            den[t] = dq + rs + 1e-6f;
            for (int s = t + 1; s < 16; ++s) att[t][s] = 0.f;
        }
        __syncthreads();
        // numerator: intra (att@v) + inter (q@S); 4 n-tiles per wave
        v16bf af_att;                       // invariant across n-tiles
        #pragma unroll
        for (int e = 0; e < 16; ++e) {
            int kk = 16 * half + e;
            af_att[e] = (__bf16)att[lane & 15][kk];
        }
        for (int i = 0; i < 4; ++i) {
            int nt = wid * 4 + i;
            int bn = nt * 16 + (lane & 15);
            v16bf bv = (half == 0) ? *(const v16bf*)(vsw + nt * 256 + (lane & 15) * 16)
                                   : vzero;
            v8f acc = {};
            acc = wmma_bf16(af_att, bv, acc);
            for (int kt = 0; kt < 16; ++kt) {
                v16bf aq = *(const v16bf*)(qs + kt * 512 + lane * 16);
                size_t tb = ((size_t)((kt * 2 + half) * 32 + nt)) * 256;
                v8f s0 = *(const v8f*)(Sb + tb + (size_t)(lane & 15) * 8);
                v8f s1 = *(const v8f*)(Sb + tb + (size_t)((lane & 15) + 16) * 8);
                v16bf bs;
                #pragma unroll
                for (int e = 0; e < 8; ++e) {
                    bs[e] = (__bf16)s0[e];
                    bs[e + 8] = (__bf16)s1[e];
                }
                acc = wmma_bf16(aq, bs, acc);
            }
            #pragma unroll
            for (int j = 0; j < 8; ++j) {
                int t = 8 * half + j;
                outB[aswz_g(rowBase + t, obCol + bn, obW)] = bf_bits(acc[j] / den[t]);
            }
        }
        __syncthreads();
        // S += k_chunk^T @ v_chunk  (rank-16, K padded; C-native tiles)
        for (int i = 0; i < 4; ++i) {
            int mt = wid * 4 + i;
            int amg = mt * 16 + (lane & 15);
            v16bf ak;
            #pragma unroll
            for (int e = 0; e < 16; ++e) {
                int kk = 16 * half + e;
                ak[e] = (kk < 16) ? bits_to_bf16(ks[kk][amg]) : (__bf16)0.f;
            }
            for (int nt = 0; nt < 32; ++nt) {
                v16bf bv = (half == 0) ? *(const v16bf*)(vsw + nt * 256 + (lane & 15) * 16)
                                       : vzero;
                float* cp = Sb + ((size_t)(mt * 32 + nt)) * 256 + (size_t)lane * 8;
                v8f cc = *(const v8f*)cp;
                cc = wmma_bf16(ak, bv, cc);
                *(v8f*)cp = cc;
            }
        }
        for (int d = tid; d < 512; d += 256) {
            float s = ksum[d];
            for (int r = 0; r < 16; ++r) s += bf2f(ks[r][d]);
            ksum[d] = s;
        }
        __syncthreads();
    }
}

// ---------------------------------------------------------------------------
// LayerNorm (+optional post-tanh); one workgroup per row
// ---------------------------------------------------------------------------
__global__ __launch_bounds__(256) void k_ln_act(
    const float* __restrict__ src, int ldS,
    float* outF, int ldF, unsigned short* outB, int obCol, int obW,
    const float* __restrict__ g, const float* __restrict__ bb,
    int dim, int postTanh) {
    __shared__ float red[256];
    __shared__ float mean_s, rstd_s;
    int row = blockIdx.x, tid = threadIdx.x;
    const float* x = src + (size_t)row * ldS;
    float s = 0.f;
    for (int d = tid; d < dim; d += 256) s += x[d];
    red[tid] = s; __syncthreads();
    for (int o = 128; o > 0; o >>= 1) { if (tid < o) red[tid] += red[tid + o]; __syncthreads(); }
    if (tid == 0) mean_s = red[0] / dim;
    __syncthreads();
    float m = mean_s;
    s = 0.f;
    for (int d = tid; d < dim; d += 256) { float t = x[d] - m; s += t * t; }
    red[tid] = s; __syncthreads();
    for (int o = 128; o > 0; o >>= 1) { if (tid < o) red[tid] += red[tid + o]; __syncthreads(); }
    if (tid == 0) rstd_s = rsqrtf(red[0] / dim + 1e-5f);
    __syncthreads();
    float rs = rstd_s;
    for (int d = tid; d < dim; d += 256) {
        float v = (x[d] - m) * rs * g[d] + bb[d];
        if (postTanh) v = tanhf(v);
        if (outF) outF[(size_t)row * ldF + d] = v;
        if (outB) outB[aswz_g(row, obCol + d, obW)] = bf_bits(v);
    }
}

__global__ __launch_bounds__(256) void k_softmax_rows(
    const float* __restrict__ src, unsigned short* __restrict__ dst, int W) {
    __shared__ float red[256];
    __shared__ float mx_s, sum_s;
    int row = blockIdx.x, tid = threadIdx.x;
    const float* x = src + (size_t)row * W;
    float m = -1e30f;
    for (int d = tid; d < W; d += 256) m = fmaxf(m, x[d]);
    red[tid] = m; __syncthreads();
    for (int o = 128; o > 0; o >>= 1) { if (tid < o) red[tid] = fmaxf(red[tid], red[tid + o]); __syncthreads(); }
    if (tid == 0) mx_s = red[0];
    __syncthreads();
    float mm = mx_s, s = 0.f;
    for (int d = tid; d < W; d += 256) s += __expf(x[d] - mm);
    red[tid] = s; __syncthreads();
    for (int o = 128; o > 0; o >>= 1) { if (tid < o) red[tid] += red[tid + o]; __syncthreads(); }
    if (tid == 0) sum_s = red[0];
    __syncthreads();
    float inv = 1.f / sum_s;
    for (int d = tid; d < W; d += 256)
        dst[aswz_g(row, d, W)] = bf_bits(__expf(x[d] - mm) * inv);
}

// ---------------------------------------------------------------------------
// Host launcher
// ---------------------------------------------------------------------------
extern "C" void kernel_launch(void* const* d_in, const int* in_sizes, int n_in,
                              void* d_out, int out_size, void* d_ws, size_t ws_size,
                              hipStream_t stream) {
    const int B = 2, T = 512, D = 512, V = 256, MEM = 256;
    const int BT = B * T, G = 3 * D;

    const int*   x        = (const int*)  d_in[0];
    const float* h_f      = (const float*)d_in[1];
    const float* h_s      = (const float*)d_in[2];
    const float* soma_w   = (const float*)d_in[3];
    const float* fast_wih = (const float*)d_in[4];
    const float* fast_whh = (const float*)d_in[5];
    const float* fast_bih = (const float*)d_in[6];
    const float* fast_bhh = (const float*)d_in[7];
    const float* slow_wih = (const float*)d_in[8];
    const float* slow_whh = (const float*)d_in[9];
    const float* slow_bih = (const float*)d_in[10];
    const float* slow_bhh = (const float*)d_in[11];
    const float* qkv_w    = (const float*)d_in[12];
    const float* qkv_b    = (const float*)d_in[13];
    const float* gate_w   = (const float*)d_in[14];
    const float* gate_b   = (const float*)d_in[15];
    const float* gate_g   = (const float*)d_in[16];
    const float* gate_bb  = (const float*)d_in[17];
    const float* mem_bank = (const float*)d_in[18];
    const float* hip_qw   = (const float*)d_in[19];
    const float* hip_qb   = (const float*)d_in[20];
    const float* axon_w   = (const float*)d_in[21];
    const float* axon_b   = (const float*)d_in[22];
    const float* norm_g   = (const float*)d_in[23];
    const float* norm_b   = (const float*)d_in[24];
    float* out = (float*)d_out;   // logits | thought | hf1 | hs1

    char* p = (char*)d_ws;
    auto aU16 = [&](size_t n) { unsigned short* r = (unsigned short*)p; p += n * 2; return r; };
    unsigned short* sin_b   = aU16((size_t)BT * D);       // A-swizzled
    unsigned short* fwihT   = aU16((size_t)D * G);        // B-swizzled
    unsigned short* fwhhSw  = aU16((size_t)D * G);
    unsigned short* swihT   = aU16((size_t)D * G);
    unsigned short* swhhSw  = aU16((size_t)D * G);
    unsigned short* qkvT    = aU16((size_t)D * G);
    unsigned short* gateT   = aU16((size_t)D * D);
    unsigned short* hipT    = aU16((size_t)D * D);
    unsigned short* axonT   = aU16((size_t)4 * D * D);
    unsigned short* memT    = aU16((size_t)D * MEM);
    unsigned short* memB    = aU16((size_t)MEM * D);
    unsigned short* swT     = aU16((size_t)D * V);
    unsigned short* pfB     = aU16((size_t)BT * D);       // A-swizzled
    unsigned short* psB     = aU16((size_t)BT * D);
    unsigned short* qhB     = aU16((size_t)BT * D);
    unsigned short* pB      = aU16((size_t)BT * MEM);
    unsigned short* thghB   = aU16((size_t)BT * D);
    unsigned short* comb    = aU16((size_t)BT * 4 * D);
    p = (char*)(((uintptr_t)p + 31) & ~(uintptr_t)31);
    auto aF = [&](size_t n) { float* r = (float*)p; p += n * 4; return r; };
    float* GIf     = aF((size_t)BT * G);
    float* GIs     = aF((size_t)BT * G);
    float* pf      = aF((size_t)BT * D);
    float* ps      = aF((size_t)BT * D);
    float* qkvF    = aF((size_t)BT * G);
    float* gatePre = aF((size_t)BT * D);
    float* scores  = aF((size_t)BT * MEM);
    float* axonA   = aF((size_t)BT * D);
    float* Sws     = aF((size_t)B * D * D);

    auto cdiv = [](int a, int b) { return (a + b - 1) / b; };
    float* out_logits  = out;
    float* out_thought = out + (size_t)BT * V;
    float* out_hf      = out + (size_t)BT * V + (size_t)BT * D;
    float* out_hs      = out_hf + (size_t)B * D;

    // ---- weight conversion into fragment-swizzled global layouts ----
    k_embed_bf16<<<cdiv(BT * D, 256), 256, 0, stream>>>(x, soma_w, sin_b, BT, D);
    k_swizzleBT_f32_bf16<<<cdiv(D * G, 256), 256, 0, stream>>>(fast_wih, fwihT, D, G);
    k_swizzleBT_f32_bf16<<<cdiv(D * G, 256), 256, 0, stream>>>(fast_whh, fwhhSw, D, G);
    k_swizzleBT_f32_bf16<<<cdiv(D * G, 256), 256, 0, stream>>>(slow_wih, swihT, D, G);
    k_swizzleBT_f32_bf16<<<cdiv(D * G, 256), 256, 0, stream>>>(slow_whh, swhhSw, D, G);
    k_swizzleBT_f32_bf16<<<cdiv(D * G, 256), 256, 0, stream>>>(qkv_w, qkvT, D, G);
    k_swizzleBT_f32_bf16<<<cdiv(D * D, 256), 256, 0, stream>>>(gate_w, gateT, D, D);
    k_swizzleBT_f32_bf16<<<cdiv(D * D, 256), 256, 0, stream>>>(hip_qw, hipT, D, D);
    k_swizzleBT_f32_bf16<<<cdiv(4 * D * D, 256), 256, 0, stream>>>(axon_w, axonT, 4 * D, D);
    k_swizzleBT_f32_bf16<<<cdiv(D * MEM, 256), 256, 0, stream>>>(mem_bank, memT, D, MEM);
    k_swizzleB_f32_bf16<<<cdiv(MEM * D, 256), 256, 0, stream>>>(mem_bank, memB, MEM, D);
    k_norm_soma<<<V, 256, 0, stream>>>(soma_w, swT, V, D);

    // ---- fast GRU: parallel input projection, then sequential recurrence ----
    k_gemm_bf16<<<dim3(G / 64, BT / 64), 256, 0, stream>>>(
        sin_b, fwihT, fast_bih, GIf, G, nullptr, 0, 0, BT, G, D, 1.f, 0, 0);
    k_gru_seq<<<1, 256, 0, stream>>>(GIf, fwhhSw, fast_bhh, h_f, pf, pfB, out_hf);

    // ---- slow GRU ----
    k_gemm_bf16<<<dim3(G / 64, BT / 64), 256, 0, stream>>>(
        pfB, swihT, slow_bih, GIs, G, nullptr, 0, 0, BT, G, D, 1.f, 0, 0);
    k_gru_seq<<<1, 256, 0, stream>>>(GIs, swhhSw, slow_bhh, h_s, ps, psB, out_hs);

    // ---- qkv (elu+1 fused on q,k columns) + chunked linear attention ----
    k_gemm_bf16<<<dim3(G / 64, BT / 64), 256, 0, stream>>>(
        psB, qkvT, qkv_b, qkvF, G, nullptr, 0, 0, BT, G, D, 1.f, 1, 2 * D);
    k_linattn<<<B, 256, 0, stream>>>(qkvF, Sws, comb, 2 * D, 4 * D);

    // ---- intent gate: linear -> LN -> tanh -> combined[:,512:1024] ----
    k_gemm_bf16<<<dim3(D / 64, BT / 64), 256, 0, stream>>>(
        psB, gateT, gate_b, gatePre, D, nullptr, 0, 0, BT, D, D, 1.f, 0, 0);
    k_ln_act<<<BT, 256, 0, stream>>>(gatePre, D, nullptr, 0, comb, D, 4 * D,
                                     gate_g, gate_bb, D, 1);

    // ---- hippocampus: qh -> scores (1/sqrt(D)) -> softmax -> episodes ----
    k_gemm_bf16<<<dim3(D / 64, BT / 64), 256, 0, stream>>>(
        psB, hipT, hip_qb, nullptr, 0, qhB, 0, D, BT, D, D, 1.f, 0, 0);
    k_gemm_bf16<<<dim3(MEM / 64, BT / 64), 256, 0, stream>>>(
        qhB, memT, nullptr, scores, MEM, nullptr, 0, 0, BT, MEM, D,
        0.044194173824159216f /* 1/sqrt(512) */, 0, 0);
    k_softmax_rows<<<BT, 256, 0, stream>>>(scores, pB, MEM);
    k_gemm_bf16<<<dim3(D / 64, BT / 64), 256, 0, stream>>>(
        pB, memB, nullptr, nullptr, 0, comb, 3 * D, 4 * D, BT, D, MEM, 1.f, 0, 0);

    // ---- combined[:,0:512] = p_f ----
    k_conv_aswz<<<cdiv(BT * D, 256), 256, 0, stream>>>(pf, D, comb, 0, 4 * D, BT, D);

    // ---- axon (tanh fused) -> LN -> thought; logits = thought @ swT * 16 ----
    k_gemm_bf16<<<dim3(D / 64, BT / 64), 256, 0, stream>>>(
        comb, axonT, axon_b, axonA, D, nullptr, 0, 0, BT, D, 4 * D, 1.f, 2, 0);
    k_ln_act<<<BT, 256, 0, stream>>>(axonA, D, out_thought, D, thghB, 0, D,
                                     norm_g, norm_b, D, 0);
    k_gemm_bf16<<<dim3(V / 64, BT / 64), 256, 0, stream>>>(
        thghB, swT, nullptr, out_logits, V, nullptr, 0, 0, BT, V, D, 16.f, 0, 0);
}
